// MoRModel_11063835754558
// MI455X (gfx1250) — compile-verified, matrix operand order checked by reading, not run on hardware
//
#include <hip/hip_runtime.h>
#include <hip/hip_bf16.h>

typedef __bf16 bf16_t;
typedef __attribute__((ext_vector_type(16))) __bf16 v16bf;
typedef __attribute__((ext_vector_type(8)))  float  v8f;
typedef __attribute__((ext_vector_type(4)))  int    v4i;

#define DMODEL 768
#define NHEAD  12
#define HD     64
#define FDIM   3072
#define NSTEP  4
#define SEQ    1024
#define BATCH  2
#define NTOK   (BATCH*SEQ)
#define VOCAB  50257

#if defined(__has_builtin)
# if __has_builtin(__builtin_amdgcn_global_load_async_to_lds_b128)
#  define HAVE_ASYNC_LDS 1
# endif
#endif

static __device__ __forceinline__ int laneId() { return (int)(threadIdx.x & 31u); }

// 16B global -> LDS copy: async (ASYNCcnt, no VGPR round trip) when available.
static __device__ __forceinline__ void cp16(bf16_t* l, const bf16_t* g) {
#ifdef HAVE_ASYNC_LDS
  __builtin_amdgcn_global_load_async_to_lds_b128(
      (__attribute__((address_space(1))) v4i*)(void*)g,
      (__attribute__((address_space(3))) v4i*)(void*)l, 0, 0);
#else
  *(uint4*)l = *(const uint4*)g;
#endif
}
static __device__ __forceinline__ void async_wait() {
#ifdef HAVE_ASYNC_LDS
  asm volatile("s_wait_asynccnt 0x0" ::: "memory");
#endif
}

// A fragment: 16x32 bf16, row-major tile (ld = row stride in elements).
// Lane L<16 holds row M=L, K in {0..7, 16..23}; lanes>=16 hold K in {8..15, 24..31}.
// Also serves as the B^T fragment loader (tile stored [N x K] row-major).
static __device__ __forceinline__ v16bf load_a_frag(const bf16_t* tile, int ld) {
  int lane = laneId(); int hlf = lane >> 4; int m = lane & 15;
  v16bf f;
#pragma unroll
  for (int e = 0; e < 16; ++e) {
    int k = 8*hlf + (e & 7) + ((e >> 3) << 4);
    f[e] = tile[m*ld + k];
  }
  return f;
}

// B fragment: 32x16 from row-major [K x N] tile (lane = column N).
static __device__ __forceinline__ v16bf load_b_frag(const bf16_t* tile, int ld) {
  int lane = laneId(); int hlf = lane >> 4; int n = lane & 15;
  v16bf f;
#pragma unroll
  for (int e = 0; e < 16; ++e) {
    int k = 8*hlf + (e & 7) + ((e >> 3) << 4);
    f[e] = tile[k*ld + n];
  }
  return f;
}

static __device__ __forceinline__ v8f wmma_bf16(v16bf a, v16bf b, v8f c) {
  return __builtin_amdgcn_wmma_f32_16x16x32_bf16(false, a, false, b, (short)0, c, false, false);
}

static __device__ __forceinline__ float gelu_tanh(float x) {
  float x3 = x*x*x;
  return 0.5f * x * (1.0f + tanhf(0.7978845608028654f * (x + 0.044715f * x3)));
}

static __device__ float blockSum(float v, float* buf) {
#pragma unroll
  for (int off = 16; off; off >>= 1) v += __shfl_xor(v, off);
  int wid = (int)(threadIdx.x >> 5);
  if ((threadIdx.x & 31u) == 0) buf[wid] = v;
  __syncthreads();
  if (threadIdx.x == 0) {
    float s = 0.f;
    int nw = (int)(blockDim.x >> 5);
    for (int i = 0; i < nw; ++i) s += buf[i];
    buf[15] = s;
  }
  __syncthreads();
  float r = buf[15];
  __syncthreads();
  return r;
}

// ---------------- elementwise / setup kernels ----------------

__global__ void castK(const float* __restrict__ in, bf16_t* __restrict__ out, int n) {
  int i = (int)(blockIdx.x*blockDim.x + threadIdx.x);
  int stride = (int)(gridDim.x*blockDim.x);
  for (; i < n; i += stride) out[i] = (bf16_t)in[i];
}

__global__ __launch_bounds__(256) void embedK(const int* __restrict__ ids,
    const float* __restrict__ tokE, const float* __restrict__ posE,
    const float* __restrict__ rw, float* __restrict__ res, int* __restrict__ depth)
{
  __shared__ float buf[16];
  int tok = (int)blockIdx.x, t = (int)threadIdx.x;
  int s = tok % SEQ;
  int id = ids[tok];
  float acc0 = 0.f, acc1 = 0.f, acc2 = 0.f, acc3 = 0.f;
  for (int j = 0; j < 3; ++j) {
    int d = t + j*256;
    float hv = tokE[(size_t)id*DMODEL + d] + posE[(size_t)s*DMODEL + d];
    res[(size_t)tok*DMODEL + d] = hv;
    acc0 += hv * rw[d*4 + 0];
    acc1 += hv * rw[d*4 + 1];
    acc2 += hv * rw[d*4 + 2];
    acc3 += hv * rw[d*4 + 3];
  }
  float l0 = blockSum(acc0, buf);
  float l1 = blockSum(acc1, buf);
  float l2 = blockSum(acc2, buf);
  float l3 = blockSum(acc3, buf);
  if (t == 0) {
    float bv = l0; int bi = 0;
    if (l1 > bv) { bv = l1; bi = 1; }
    if (l2 > bv) { bv = l2; bi = 2; }
    if (l3 > bv) { bv = l3; bi = 3; }
    depth[tok] = bi + 1;
  }
}

__global__ __launch_bounds__(256) void lnK(const float* __restrict__ in,
    const float* __restrict__ scale, const float* __restrict__ bias,
    bf16_t* __restrict__ out)
{
  __shared__ float buf[16];
  int tok = (int)blockIdx.x, t = (int)threadIdx.x;
  float vals[3];
  for (int j = 0; j < 3; ++j) vals[j] = in[(size_t)tok*DMODEL + t + j*256];
  float s = vals[0] + vals[1] + vals[2];
  float mean = blockSum(s, buf) * (1.0f/DMODEL);
  float q = 0.f;
  for (int j = 0; j < 3; ++j) { float d = vals[j] - mean; q += d*d; }
  float var = blockSum(q, buf) * (1.0f/DMODEL);
  float rstd = rsqrtf(var + 1e-5f);
  for (int j = 0; j < 3; ++j) {
    int dcol = t + j*256;
    out[(size_t)tok*DMODEL + dcol] = (bf16_t)((vals[j]-mean)*rstd*scale[dcol] + bias[dcol]);
  }
}

__global__ void depthCopyK(const int* __restrict__ depth, int* __restrict__ out) {
  int i = (int)(blockIdx.x*blockDim.x + threadIdx.x);
  if (i < NTOK) out[i] = depth[i];
}

// ---------------- generic bf16 WMMA GEMM ----------------
// C[M,N] = A[M,K] x B[K,N]; 128x128 tile per WG (256 thr, 8 waves, 32x64 per wave),
// K-step 32, double-buffered LDS, async global->LDS staging.
// EPI: 1=store bf16  2=f32 +res  3=+bias,gelu,bf16  4=gated residual update  5=logits (N-guard)

template<bool BT>
static __device__ __forceinline__ void stage_tiles(
    const bf16_t* __restrict__ A, int lda, const bf16_t* __restrict__ B, int ldb,
    int m0, int n0, int k0, int N, bf16_t* lA, bf16_t* lB, int t)
{
  for (int c = t; c < 512; c += 256) {               // A: 128 rows x 32 halves
    int r = c >> 2, q = c & 3;
    cp16(lA + r*32 + q*8, A + (size_t)(m0 + r)*lda + k0 + q*8);
  }
  if (BT) {                                           // B stored [N x K]: 128 n-rows x 32 halves
    for (int c = t; c < 512; c += 256) {
      int r = c >> 2, q = c & 3;
      int gn = n0 + r;
      if (gn < N) cp16(lB + r*32 + q*8, B + (size_t)gn*ldb + k0 + q*8);
      else { uint4 z = {0u,0u,0u,0u}; *(uint4*)(lB + r*32 + q*8) = z; }
    }
  } else {                                            // B: 32 k-rows x 128 halves
    for (int c = t; c < 512; c += 256) {
      int r = c >> 4, q = c & 15;
      cp16(lB + r*128 + q*8, B + (size_t)(k0 + r)*ldb + n0 + q*8);
    }
  }
}

template<int EPI, bool BT>
__global__ __launch_bounds__(256) void gemmK(const bf16_t* __restrict__ A, int lda,
    const bf16_t* __restrict__ B, int ldb, int N, int K,
    float* __restrict__ outF, bf16_t* __restrict__ outB, int ldo,
    const float* __restrict__ bias, const float* __restrict__ resIn,
    float* __restrict__ resOut, const int* __restrict__ depth, int step)
{
  __shared__ __align__(16) bf16_t lA[2][128*32];
  __shared__ __align__(16) bf16_t lB[2][128*32];
  int m0 = (int)blockIdx.y * 128;
  int n0 = (int)blockIdx.x * 128;
  int t = (int)threadIdx.x;
  int wv = t >> 5;
  int mi = (wv & 3) * 32;    // wave's row offset in tile (2 M-frags)
  int ni = (wv >> 2) * 64;   // wave's col offset in tile (4 N-frags)
  v8f acc[2][4] = {};
  int nk = K / 32;

  stage_tiles<BT>(A, lda, B, ldb, m0, n0, 0, N, lA[0], lB[0], t);
  async_wait();
  __syncthreads();

  for (int ks = 0; ks < nk; ++ks) {
    int cur = ks & 1, nxt = cur ^ 1;
    if (ks + 1 < nk)
      stage_tiles<BT>(A, lda, B, ldb, m0, n0, (ks+1)*32, N, lA[nxt], lB[nxt], t);

    v16bf a0 = load_a_frag(lA[cur] + (size_t)mi*32, 32);
    v16bf a1 = load_a_frag(lA[cur] + (size_t)(mi+16)*32, 32);
#pragma unroll
    for (int nt = 0; nt < 4; ++nt) {
      v16bf b = BT ? load_a_frag(lB[cur] + (size_t)(ni + nt*16)*32, 32)
                   : load_b_frag(lB[cur] + ni + nt*16, 128);
      acc[0][nt] = wmma_bf16(a0, b, acc[0][nt]);
      acc[1][nt] = wmma_bf16(a1, b, acc[1][nt]);
    }
    async_wait();
    __syncthreads();
  }

  int lane = laneId(); int hlf = lane >> 4; int nn = lane & 15;
#pragma unroll
  for (int mt = 0; mt < 2; ++mt) {
#pragma unroll
    for (int nt = 0; nt < 4; ++nt) {
#pragma unroll
      for (int v = 0; v < 8; ++v) {
        int gm = m0 + mi + mt*16 + v + 8*hlf;
        int gn = n0 + ni + nt*16 + nn;
        float cv = acc[mt][nt][v];
        if (EPI == 1) {
          outB[(size_t)gm*ldo + gn] = (bf16_t)cv;
        } else if (EPI == 2) {
          outF[(size_t)gm*ldo + gn] = resIn[(size_t)gm*ldo + gn] + cv;
        } else if (EPI == 3) {
          outB[(size_t)gm*ldo + gn] = (bf16_t)gelu_tanh(cv + bias[gn]);
        } else if (EPI == 4) {
          if (depth[gm] >= step + 1)
            resOut[(size_t)gm*ldo + gn] = resIn[(size_t)gm*ldo + gn] + cv + bias[gn];
        } else { // 5: logits
          if (gn < N) outF[(size_t)gm*ldo + gn] = cv;
        }
      }
    }
  }
}

// ---------------- flash attention with MoR masking ----------------
// grid: (S/64, H, B), 128 threads (4 waves, each owns 16 q rows).
__global__ __launch_bounds__(128) void attnK(const bf16_t* __restrict__ qkv,
    const int* __restrict__ depth, bf16_t* __restrict__ outA, int step)
{
  __shared__ __align__(16) bf16_t lK[32*64];
  __shared__ __align__(16) bf16_t lV[32*64];
  __shared__ __align__(16) bf16_t lP[4][16*32];
  int bb = (int)blockIdx.z, h = (int)blockIdx.y, qblk = (int)blockIdx.x;
  int t = (int)threadIdx.x, wv = t >> 5, lane = laneId();
  int hlf = lane >> 4, nn = lane & 15;
  int qrow0 = qblk*64 + wv*16;

  const bf16_t* qptr = qkv + ((size_t)(bb*SEQ + qrow0))*(3*DMODEL) + h*HD;
  v16bf qf0 = load_a_frag(qptr, 3*DMODEL);        // Q dims 0..31
  v16bf qf1 = load_a_frag(qptr + 32, 3*DMODEL);   // Q dims 32..63

  float run_max[8], run_sum[8], alpha[8];
  v8f o[4] = {};
#pragma unroll
  for (int v = 0; v < 8; ++v) { run_max[v] = -3.0e38f; run_sum[v] = 0.f; }

  int nblk = (qblk + 1) * 2;                       // causal: keys up to q-tile end
  for (int j = 0; j < nblk; ++j) {
    int jbase = j * 32;
    __syncthreads();
    for (int c = t; c < 256; c += 128) {           // stage K/V block (32 keys x 64 dims)
      int r = c >> 3, q = c & 7;
      size_t g = ((size_t)(bb*SEQ + jbase + r))*(3*DMODEL) + DMODEL + h*HD + q*8;
      cp16(lK + r*64 + q*8, qkv + g);
      cp16(lV + r*64 + q*8, qkv + g + DMODEL);
    }
    async_wait();
    __syncthreads();

    v8f sc[2] = {};
#pragma unroll
    for (int nt = 0; nt < 2; ++nt) {               // S = Q(16x64) x K^T(64x16), per 16 keys
      v16bf b0 = load_a_frag(lK + nt*16*64, 64);
      v16bf b1 = load_a_frag(lK + nt*16*64 + 32, 64);
      sc[nt] = wmma_bf16(qf0, b0, sc[nt]);
      sc[nt] = wmma_bf16(qf1, b1, sc[nt]);
    }

    float pv[2][8];
#pragma unroll
    for (int nt = 0; nt < 2; ++nt) {
      int kidx = jbase + nt*16 + nn;
      float kb = (depth[bb*SEQ + kidx] >= step + 1) ? 0.f : -1e9f;
#pragma unroll
      for (int v = 0; v < 8; ++v) {
        int qidx = qrow0 + v + 8*hlf;
        float mval = ((kidx > qidx) ? -1e9f : 0.f) + kb;  // causal + key bias
        if (kidx == qidx) mval = 0.f;                     // diagonal always open
        pv[nt][v] = sc[nt][v] * 0.125f + mval;            // 1/sqrt(64)
      }
    }
#pragma unroll
    for (int v = 0; v < 8; ++v) {                  // online softmax, per q row
      float mx = fmaxf(pv[0][v], pv[1][v]);
#pragma unroll
      for (int off = 8; off; off >>= 1) mx = fmaxf(mx, __shfl_xor(mx, off));
      float nm = fmaxf(run_max[v], mx);
      alpha[v] = __expf(run_max[v] - nm);
      run_max[v] = nm;
      float p0 = __expf(pv[0][v] - nm);
      float p1 = __expf(pv[1][v] - nm);
      pv[0][v] = p0; pv[1][v] = p1;
      float ls = p0 + p1;
#pragma unroll
      for (int off = 8; off; off >>= 1) ls += __shfl_xor(ls, off);
      run_sum[v] = run_sum[v]*alpha[v] + ls;
    }
#pragma unroll
    for (int tt = 0; tt < 4; ++tt)
#pragma unroll
      for (int v = 0; v < 8; ++v) o[tt][v] *= alpha[v];

#pragma unroll
    for (int nt = 0; nt < 2; ++nt)                 // P -> LDS (C layout -> A layout via LDS)
#pragma unroll
      for (int v = 0; v < 8; ++v)
        lP[wv][(v + 8*hlf)*32 + nt*16 + nn] = (bf16_t)pv[nt][v];
    __syncthreads();

    v16bf pa = load_a_frag(lP[wv], 32);
#pragma unroll
    for (int tt = 0; tt < 4; ++tt) {               // O += P(16x32) x V(32x64)
      v16bf vb = load_b_frag(lV + tt*16, 64);
      o[tt] = wmma_bf16(pa, vb, o[tt]);
    }
  }
#pragma unroll
  for (int tt = 0; tt < 4; ++tt)
#pragma unroll
    for (int v = 0; v < 8; ++v) {
      float val = o[tt][v] / run_sum[v];
      size_t idx = ((size_t)(bb*SEQ + qrow0 + v + 8*hlf))*DMODEL + h*HD + tt*16 + nn;
      outA[idx] = (bf16_t)val;
    }
}

// ---------------- host orchestration ----------------

extern "C" void kernel_launch(void* const* d_in, const int* in_sizes, int n_in,
                              void* d_out, int out_size, void* d_ws, size_t ws_size,
                              hipStream_t stream) {
  const int*   ids  = (const int*)d_in[0];
  const float* tokE = (const float*)d_in[1];
  const float* posE = (const float*)d_in[2];
  const float* rw   = (const float*)d_in[3];
  const float* ln1s = (const float*)d_in[4];
  const float* ln1b = (const float*)d_in[5];
  const float* wqkv = (const float*)d_in[6];
  const float* wo   = (const float*)d_in[7];
  const float* ln2s = (const float*)d_in[8];
  const float* ln2b = (const float*)d_in[9];
  const float* w1   = (const float*)d_in[10];
  const float* b1   = (const float*)d_in[11];
  const float* w2   = (const float*)d_in[12];
  const float* b2   = (const float*)d_in[13];
  const float* lnfs = (const float*)d_in[14];
  const float* lnfb = (const float*)d_in[15];

  char* p = (char*)d_ws;
  auto carve = [&](size_t bytes) { char* r = p; p += (bytes + 255) & ~(size_t)255; return r; };
  float*  res   = (float*)  carve((size_t)NTOK*DMODEL*4);
  float*  x1    = (float*)  carve((size_t)NTOK*DMODEL*4);
  bf16_t* xbf   = (bf16_t*) carve((size_t)NTOK*DMODEL*2);
  bf16_t* qkvbf = (bf16_t*) carve((size_t)NTOK*3*DMODEL*2);
  bf16_t* attbf = (bf16_t*) carve((size_t)NTOK*DMODEL*2);
  bf16_t* y1bf  = (bf16_t*) carve((size_t)NTOK*FDIM*2);
  int*    depth = (int*)    carve((size_t)NTOK*4);
  bf16_t* wqkvb = (bf16_t*) carve((size_t)DMODEL*3*DMODEL*2);
  bf16_t* wob   = (bf16_t*) carve((size_t)DMODEL*DMODEL*2);
  bf16_t* w1b   = (bf16_t*) carve((size_t)DMODEL*FDIM*2);
  bf16_t* w2b   = (bf16_t*) carve((size_t)FDIM*DMODEL*2);
  bf16_t* embb  = (bf16_t*) carve((size_t)VOCAB*DMODEL*2);

  castK<<<1024, 256, 0, stream>>>(wqkv, wqkvb, DMODEL*3*DMODEL);
  castK<<<1024, 256, 0, stream>>>(wo,   wob,   DMODEL*DMODEL);
  castK<<<1024, 256, 0, stream>>>(w1,   w1b,   DMODEL*FDIM);
  castK<<<1024, 256, 0, stream>>>(w2,   w2b,   FDIM*DMODEL);
  castK<<<4096, 256, 0, stream>>>(tokE, embb,  VOCAB*DMODEL);

  embedK<<<NTOK, 256, 0, stream>>>(ids, tokE, posE, rw, res, depth);

  dim3 gQKV(3*DMODEL/128, NTOK/128);
  dim3 gO(DMODEL/128, NTOK/128);
  dim3 gF(FDIM/128, NTOK/128);
  dim3 gAttn(SEQ/64, NHEAD, BATCH);

  for (int step = 0; step < NSTEP; ++step) {
    lnK<<<NTOK, 256, 0, stream>>>(res, ln1s, ln1b, xbf);
    gemmK<1,false><<<gQKV, 256, 0, stream>>>(xbf, DMODEL, wqkvb, 3*DMODEL, 3*DMODEL, DMODEL,
        nullptr, qkvbf, 3*DMODEL, nullptr, nullptr, nullptr, nullptr, 0);
    attnK<<<gAttn, 128, 0, stream>>>(qkvbf, depth, attbf, step);
    gemmK<2,false><<<gO, 256, 0, stream>>>(attbf, DMODEL, wob, DMODEL, DMODEL, DMODEL,
        x1, nullptr, DMODEL, nullptr, res, nullptr, nullptr, 0);
    lnK<<<NTOK, 256, 0, stream>>>(x1, ln2s, ln2b, xbf);
    gemmK<3,false><<<gF, 256, 0, stream>>>(xbf, DMODEL, w1b, FDIM, FDIM, DMODEL,
        nullptr, y1bf, FDIM, b1, nullptr, nullptr, nullptr, 0);
    gemmK<4,false><<<gO, 256, 0, stream>>>(y1bf, FDIM, w2b, DMODEL, DMODEL, FDIM,
        nullptr, nullptr, DMODEL, b2, x1, res, depth, step);
  }

  lnK<<<NTOK, 256, 0, stream>>>(res, lnfs, lnfb, xbf);
  float* logits = (float*)d_out;
  dim3 gL((VOCAB + 127)/128, NTOK/128);
  gemmK<5,true><<<gL, 256, 0, stream>>>(xbf, DMODEL, embb, DMODEL, VOCAB, DMODEL,
      logits, nullptr, VOCAB, nullptr, nullptr, nullptr, nullptr, 0);
  depthCopyK<<<(NTOK+255)/256, 256, 0, stream>>>(depth, (int*)(logits + (size_t)NTOK*VOCAB));
}
